// VideoMultiscale_Deform2d_87316685128182
// MI455X (gfx1250) — matrix operand order repaired and checked by reading, not compiled, not run on hardware
//
#include <hip/hip_runtime.h>
#include <math.h>

// ---------------------------------------------------------------------------
// MI455X / gfx1250 implementation of a 2-layer Deformable-DETR encoder.
// GEMMs run on v_wmma_f32_16x16x32_bf16, one wave = 32x32 C tile (2x2 WMMA),
// with an explicit 2-stage software pipeline (load next frags before current
// WMMAs) so the loop waits only on the previous iteration's loads.
// ---------------------------------------------------------------------------

typedef __bf16 bf16_t;
typedef __attribute__((ext_vector_type(16))) __bf16 v16bf;
typedef __attribute__((ext_vector_type(8)))  float  v8f;

#define BATCH  8
#define SEQ    5440           // LEN == Q
#define DM     256
#define NH     8
#define HD     32
#define DFF    1024
#define NLAYER 2
#define ROWS   (BATCH * SEQ)  // 43520, divisible by 32

union FragAB { v16bf bf; uint4 u[2]; };

static __device__ __forceinline__ unsigned short f2bf_raw(float f) {
  bf16_t h = (bf16_t)f;            // RNE f32->bf16
  unsigned short u;
  __builtin_memcpy(&u, &h, 2);
  return u;
}

// level of a flattened token index (64x64, 32x32, 16x16, 8x8)
static __device__ __forceinline__ int tok_level(int q, int* rloc) {
  if (q < 4096)      { *rloc = q;        return 0; }
  else if (q < 5120) { *rloc = q - 4096; return 1; }
  else if (q < 5376) { *rloc = q - 5120; return 2; }
  else               { *rloc = q - 5376; return 3; }
}

// qpos = pos + level_embed[level(q)]
__global__ void qpos_kernel(const float* __restrict__ pos,
                            const float* __restrict__ lev,
                            float* __restrict__ qpos) {
  size_t i = (size_t)blockIdx.x * blockDim.x + threadIdx.x;
  if (i >= (size_t)ROWS * DM) return;
  int c = (int)(i & (DM - 1));
  int q = (int)((i >> 8) % SEQ);
  int r;
  int l = tok_level(q, &r);
  qpos[i] = pos[i] + lev[l * DM + c];
}

// qbf = bf16(x + qpos); xbf = bf16(x)
__global__ void prep_kernel(const float* __restrict__ x,
                            const float* __restrict__ qpos,
                            bf16_t* __restrict__ qbf,
                            bf16_t* __restrict__ xbf) {
  size_t i = (size_t)blockIdx.x * blockDim.x + threadIdx.x;
  if (i >= (size_t)ROWS * DM) return;
  float xv = x[i];
  qbf[i] = (bf16_t)(xv + qpos[i]);
  xbf[i] = (bf16_t)xv;
}

__global__ void cast_kernel(const float* __restrict__ in,
                            bf16_t* __restrict__ out, size_t n) {
  size_t i = (size_t)blockIdx.x * blockDim.x + threadIdx.x;
  if (i < n) out[i] = (bf16_t)in[i];
}

// Pack row-major f32 W[K][N] into WMMA B-fragment order:
// dword id = ((nTile16*(K/32)+kt)*256 + lane*8 + v)
// lane: n = nTile16*16 + (lane&15), half = lane>>4 ; K pair = kt*32 + (v>>2)*16 + half*8 + (v&3)*2
__global__ void pack_w_kernel(const float* __restrict__ W,
                              unsigned int* __restrict__ Wp, int K, int N) {
  int id = blockIdx.x * blockDim.x + threadIdx.x;
  int total = (K * N) >> 1;
  if (id >= total) return;
  int v    = id & 7;
  int lane = (id >> 3) & 31;
  int rest = id >> 8;
  int kIter = K >> 5;
  int kt = rest % kIter;
  int nT = rest / kIter;
  int n    = nT * 16 + (lane & 15);
  int half = lane >> 4;
  int k    = kt * 32 + (v >> 2) * 16 + half * 8 + (v & 3) * 2;
  unsigned int lo = f2bf_raw(W[(size_t)k * N + n]);
  unsigned int hi = f2bf_raw(W[(size_t)(k + 1) * N + n]);
  Wp[id] = lo | (hi << 16);
}

static __device__ __forceinline__ void store_tile(
    v8f c, float bv, int mbase, int ncol, int N, int relu, int out_bf,
    float* __restrict__ Cf, bf16_t* __restrict__ Cb) {
#pragma unroll
  for (int r = 0; r < 8; ++r) {
    float y = c[r] + bv;
    if (relu) y = fmaxf(y, 0.f);
    size_t idx = (size_t)(mbase + r) * N + ncol;
    if (out_bf) Cb[idx] = (bf16_t)y;
    else        Cf[idx] = y;
  }
}

// C[M,N] = A[M,K](bf16 row-major) @ Wp(packed) + bias ; optional relu / bf16 out
// one wave -> one 32x32 tile (2x2 of 16x16 WMMA), 8 waves per block
__global__ void __launch_bounds__(256) gemm_bf16_wmma(
    const bf16_t* __restrict__ A, const unsigned int* __restrict__ Wp,
    const float* __restrict__ bias, float* __restrict__ Cf,
    bf16_t* __restrict__ Cb, int K, int N, int relu, int out_bf) {
  int lane = threadIdx.x & 31;
  int wv   = threadIdx.x >> 5;
  int tile = blockIdx.x * 8 + wv;
  int nT32 = N >> 5;
  int mT = tile / nT32;
  int nT = tile - mT * nT32;
  int half = lane >> 4;
  int m0 = mT * 32;
  int n0 = nT * 32;
  int kIter = K >> 5;

  // A lane layout (16-bit A 16x32): V0-3 = K[half*8 .. +7], V4-7 = K[16+half*8 .. +7]
  const bf16_t* ap0 = A + (size_t)(m0 + (lane & 15)) * K + half * 8;
  const bf16_t* ap1 = ap0 + (size_t)16 * K;
  // B: pre-packed 16-col tile blocks; 8 consecutive dwords per lane per k-step
  const uint4* bp0 = (const uint4*)(Wp + (size_t)(2 * nT) * kIter * 256) + lane * 2;
  const uint4* bp1 = bp0 + (size_t)kIter * 64;

  v8f c00 = {0.f, 0.f, 0.f, 0.f, 0.f, 0.f, 0.f, 0.f};
  v8f c01 = c00, c10 = c00, c11 = c00;

  // prologue: stage k-step 0
  FragAB a0, a1, b0, b1;
  a0.u[0] = *(const uint4*)(ap0);
  a0.u[1] = *(const uint4*)(ap0 + 16);
  a1.u[0] = *(const uint4*)(ap1);
  a1.u[1] = *(const uint4*)(ap1 + 16);
  b0.u[0] = bp0[0];
  b0.u[1] = bp0[1];
  b1.u[0] = bp1[0];
  b1.u[1] = bp1[1];

  for (int kt = 0; kt < kIter - 1; ++kt) {
    ap0 += 32; ap1 += 32;
    bp0 += 64; bp1 += 64;   // 256 dwords per k-step
    // issue next iteration's loads first (disjoint regs -> partial loadcnt wait)
    FragAB na0, na1, nb0, nb1;
    na0.u[0] = *(const uint4*)(ap0);
    na0.u[1] = *(const uint4*)(ap0 + 16);
    na1.u[0] = *(const uint4*)(ap1);
    na1.u[1] = *(const uint4*)(ap1 + 16);
    nb0.u[0] = bp0[0];
    nb0.u[1] = bp0[1];
    nb1.u[0] = bp1[0];
    nb1.u[1] = bp1[1];
    __builtin_prefetch(ap0 + 32, 0, 3);   // prefetch k-step kt+2 A chunks
    __builtin_prefetch(ap1 + 32, 0, 3);
    // consume current fragments (independent accumulators: no WMMA RAW chain)
    c00 = __builtin_amdgcn_wmma_f32_16x16x32_bf16(
        false, a0.bf, false, b0.bf, (short)0, c00, false, false);
    c01 = __builtin_amdgcn_wmma_f32_16x16x32_bf16(
        false, a0.bf, false, b1.bf, (short)0, c01, false, false);
    c10 = __builtin_amdgcn_wmma_f32_16x16x32_bf16(
        false, a1.bf, false, b0.bf, (short)0, c10, false, false);
    c11 = __builtin_amdgcn_wmma_f32_16x16x32_bf16(
        false, a1.bf, false, b1.bf, (short)0, c11, false, false);
    a0 = na0; a1 = na1; b0 = nb0; b1 = nb1;
  }
  // epilogue k-step
  c00 = __builtin_amdgcn_wmma_f32_16x16x32_bf16(
      false, a0.bf, false, b0.bf, (short)0, c00, false, false);
  c01 = __builtin_amdgcn_wmma_f32_16x16x32_bf16(
      false, a0.bf, false, b1.bf, (short)0, c01, false, false);
  c10 = __builtin_amdgcn_wmma_f32_16x16x32_bf16(
      false, a1.bf, false, b0.bf, (short)0, c10, false, false);
  c11 = __builtin_amdgcn_wmma_f32_16x16x32_bf16(
      false, a1.bf, false, b1.bf, (short)0, c11, false, false);

  // C layout per 16x16 tile: lanes 0-15: M = base + r, N = lane ; lanes 16-31: M += 8
  int nc0 = n0 + (lane & 15);
  int nc1 = nc0 + 16;
  int mb0 = m0 + half * 8;
  int mb1 = mb0 + 16;
  float bv0 = bias[nc0];
  float bv1 = bias[nc1];
  store_tile(c00, bv0, mb0, nc0, N, relu, out_bf, Cf, Cb);
  store_tile(c01, bv1, mb0, nc1, N, relu, out_bf, Cf, Cb);
  store_tile(c10, bv0, mb1, nc0, N, relu, out_bf, Cf, Cb);
  store_tile(c11, bv1, mb1, nc1, N, relu, out_bf, Cf, Cb);
}

// softmax over 16 contiguous values per (b,q,head)
__global__ void softmax16_kernel(float* __restrict__ aw) {
  int i = blockIdx.x * blockDim.x + threadIdx.x;
  if (i >= ROWS * NH) return;
  float* p = aw + (size_t)i * 16;
  float v[16];
  float m = -1e30f;
#pragma unroll
  for (int j = 0; j < 16; ++j) { v[j] = p[j]; m = fmaxf(m, v[j]); }
  float s = 0.f;
#pragma unroll
  for (int j = 0; j < 16; ++j) { v[j] = __expf(v[j] - m); s += v[j]; }
  float inv = 1.f / s;
#pragma unroll
  for (int j = 0; j < 16; ++j) p[j] = v[j] * inv;
}

// multi-scale deformable sampling: one wave per (b,q,head); lane = channel
__global__ void __launch_bounds__(256) deform_kernel(
    const float* __restrict__ value, const float* __restrict__ off,
    const float* __restrict__ aw, float* __restrict__ samp) {
  int lane = threadIdx.x & 31;
  int h    = threadIdx.x >> 5;
  int bq   = blockIdx.x;            // gridDim.x == ROWS
  int b    = bq / SEQ;
  int q    = bq - b * SEQ;
  int r;
  int lq = tok_level(q, &r);
  int Wq = 64 >> lq;
  float refx = ((r % Wq) + 0.5f) / (float)Wq;
  float refy = ((r / Wq) + 0.5f) / (float)Wq;

  const float* offp = off + (size_t)bq * 256 + h * 32;  // (l*4+p)*2 + {x,y}
  const float* awp  = aw  + (size_t)bq * 128 + h * 16;  // l*4+p
  const int starts[4] = {0, 4096, 5120, 5376};

  float acc = 0.f;
#pragma unroll
  for (int l = 0; l < 4; ++l) {
    int Wl = 64 >> l;
    float fW = (float)Wl;
    size_t vbase = ((size_t)b * SEQ + starts[l]) * DM + h * HD + lane;
#pragma unroll
    for (int p = 0; p < 4; ++p) {
      float ox = offp[(l * 4 + p) * 2 + 0];
      float oy = offp[(l * 4 + p) * 2 + 1];
      float a  = awp[l * 4 + p];
      // loc*W - 0.5 = ref*W + off - 0.5  (NORM divides off, grid mult cancels)
      float gx = refx * fW + ox - 0.5f;
      float gy = refy * fW + oy - 0.5f;
      float x0f = floorf(gx), y0f = floorf(gy);
      int   x0  = (int)x0f,   y0  = (int)y0f;
      float wx1 = gx - x0f,   wy1 = gy - y0f;
      float wx0 = 1.f - wx1,  wy0 = 1.f - wy1;
#pragma unroll
      for (int t = 0; t < 4; ++t) {
        int dx = t & 1, dy = t >> 1;
        int xi = x0 + dx, yi = y0 + dy;
        if (xi >= 0 && xi < Wl && yi >= 0 && yi < Wl) {   // zero padding
          float w  = (dx ? wx1 : wx0) * (dy ? wy1 : wy0);
          float vv = value[vbase + (size_t)(yi * Wl + xi) * DM];
          acc += a * w * vv;
        }
      }
    }
  }
  samp[(size_t)bq * DM + h * HD + lane] = acc;
}

// xout = LayerNorm(xin + delta) * g + b ; optional bf16 copy. One wave per row.
__global__ void __launch_bounds__(256) ln_kernel(
    const float* __restrict__ xin, const float* __restrict__ delta,
    const float* __restrict__ g, const float* __restrict__ bta,
    float* __restrict__ xout, bf16_t* __restrict__ xbf) {
  int lane = threadIdx.x & 31;
  int row  = blockIdx.x * 8 + (threadIdx.x >> 5);
  const float* xi = xin   + (size_t)row * DM;
  const float* dl = delta + (size_t)row * DM;
  float v[8];
  float s = 0.f, s2 = 0.f;
#pragma unroll
  for (int j = 0; j < 8; ++j) {
    float t = xi[lane + 32 * j] + dl[lane + 32 * j];
    v[j] = t; s += t; s2 += t * t;
  }
#pragma unroll
  for (int m = 16; m > 0; m >>= 1) {   // wave32 reduction
    s  += __shfl_xor(s,  m, 32);
    s2 += __shfl_xor(s2, m, 32);
  }
  float mean = s * (1.f / DM);
  float var  = s2 * (1.f / DM) - mean * mean;
  float inv  = rsqrtf(var + 1e-5f);
#pragma unroll
  for (int j = 0; j < 8; ++j) {
    int cc = lane + 32 * j;
    float y = (v[j] - mean) * inv * g[cc] + bta[cc];
    xout[(size_t)row * DM + cc] = y;
    if (xbf) xbf[(size_t)row * DM + cc] = (bf16_t)y;
  }
}

extern "C" void kernel_launch(void* const* d_in, const int* in_sizes, int n_in,
                              void* d_out, int out_size, void* d_ws, size_t ws_size,
                              hipStream_t stream) {
  (void)in_sizes; (void)n_in; (void)out_size; (void)ws_size;

  const float* src    = (const float*)d_in[0];
  const float* pos    = (const float*)d_in[1];
  const float* lev    = (const float*)d_in[2];
  const float* w_off  = (const float*)d_in[3];
  const float* b_off  = (const float*)d_in[4];
  const float* w_attn = (const float*)d_in[5];
  const float* b_attn = (const float*)d_in[6];
  const float* w_val  = (const float*)d_in[7];
  const float* b_val  = (const float*)d_in[8];
  const float* w_out  = (const float*)d_in[9];
  const float* b_out  = (const float*)d_in[10];
  const float* ln1g   = (const float*)d_in[11];
  const float* ln1b   = (const float*)d_in[12];
  const float* w_ff1  = (const float*)d_in[13];
  const float* b_ff1  = (const float*)d_in[14];
  const float* w_ff2  = (const float*)d_in[15];
  const float* b_ff2  = (const float*)d_in[16];
  const float* ln2g   = (const float*)d_in[17];
  const float* ln2b   = (const float*)d_in[18];

  char* ws = (char*)d_ws;
  size_t o = 0;
  auto carve = [&](size_t bytes) {
    size_t r = o; o += (bytes + 255) & ~(size_t)255; return r;
  };
  float*  qpos = (float*)(ws + carve((size_t)ROWS * DM * 4));
  float*  x    = (float*)(ws + carve((size_t)ROWS * DM * 4));
  float*  x2   = (float*)(ws + carve((size_t)ROWS * DM * 4));  // also samp
  float*  offb = (float*)(ws + carve((size_t)ROWS * DM * 4));  // off -> attn out
  float*  awb  = (float*)(ws + carve((size_t)ROWS * 128 * 4));
  float*  valb = (float*)(ws + carve((size_t)ROWS * DM * 4));  // value -> ff2 out
  bf16_t* qbf  = (bf16_t*)(ws + carve((size_t)ROWS * DM * 2)); // q -> samp_bf
  bf16_t* abf  = (bf16_t*)(ws + carve((size_t)ROWS * DM * 2)); // x_bf -> x2_bf
  bf16_t* hbf  = (bf16_t*)(ws + carve((size_t)ROWS * DFF * 2));
  unsigned int* wp_off  = (unsigned int*)(ws + carve(32768u * 4));
  unsigned int* wp_attn = (unsigned int*)(ws + carve(16384u * 4));
  unsigned int* wp_val  = (unsigned int*)(ws + carve(32768u * 4));
  unsigned int* wp_out  = (unsigned int*)(ws + carve(32768u * 4));
  unsigned int* wp_ff1  = (unsigned int*)(ws + carve(131072u * 4));
  unsigned int* wp_ff2  = (unsigned int*)(ws + carve(131072u * 4));

  const size_t nElem = (size_t)ROWS * DM;     // 11,141,120
  const int    eBlk  = 43520;                 // nElem / 256

  hipMemcpyAsync(x, src, nElem * 4, hipMemcpyDeviceToDevice, stream);
  qpos_kernel<<<eBlk, 256, 0, stream>>>(pos, lev, qpos);

  for (int i = 0; i < NLAYER; ++i) {
    // pack weights into WMMA B-fragment order
    pack_w_kernel<<<128, 256, 0, stream>>>(w_off  + (size_t)i * DM * 256, wp_off,  256, 256);
    pack_w_kernel<<< 64, 256, 0, stream>>>(w_attn + (size_t)i * DM * 128, wp_attn, 256, 128);
    pack_w_kernel<<<128, 256, 0, stream>>>(w_val  + (size_t)i * DM * DM,  wp_val,  256, 256);
    pack_w_kernel<<<128, 256, 0, stream>>>(w_out  + (size_t)i * DM * DM,  wp_out,  256, 256);
    pack_w_kernel<<<512, 256, 0, stream>>>(w_ff1  + (size_t)i * DM * DFF, wp_ff1,  256, 1024);
    pack_w_kernel<<<512, 256, 0, stream>>>(w_ff2  + (size_t)i * DFF * DM, wp_ff2,  1024, 256);

    prep_kernel<<<eBlk, 256, 0, stream>>>(x, qpos, qbf, abf);

    // sampling-offset / attention-logit / value projections (bf16 WMMA)
    // blocks = (M/32)*(N/32)/8 waves
    gemm_bf16_wmma<<<1360, 256, 0, stream>>>(qbf, wp_off,  b_off  + i * 256, offb, nullptr, 256, 256, 0, 0);
    gemm_bf16_wmma<<< 680, 256, 0, stream>>>(qbf, wp_attn, b_attn + i * 128, awb,  nullptr, 256, 128, 0, 0);
    gemm_bf16_wmma<<<1360, 256, 0, stream>>>(abf, wp_val,  b_val  + i * 256, valb, nullptr, 256, 256, 0, 0);

    softmax16_kernel<<<(ROWS * NH + 255) / 256, 256, 0, stream>>>(awb);
    deform_kernel<<<ROWS, 256, 0, stream>>>(valb, offb, awb, x2 /*samp*/);

    cast_kernel<<<eBlk, 256, 0, stream>>>(x2, qbf, nElem);
    gemm_bf16_wmma<<<1360, 256, 0, stream>>>(qbf, wp_out, b_out + i * 256, offb /*attn*/, nullptr, 256, 256, 0, 0);

    ln_kernel<<<5440, 256, 0, stream>>>(x, offb, ln1g + i * 256, ln1b + i * 256, x2, abf);

    // FFN: relu GEMM writing bf16 directly, then down-projection
    gemm_bf16_wmma<<<5440, 256, 0, stream>>>(abf, wp_ff1, b_ff1 + i * DFF, nullptr, hbf, 256, 1024, 1, 1);
    gemm_bf16_wmma<<<1360, 256, 0, stream>>>(hbf, wp_ff2, b_ff2 + i * 256, valb, nullptr, 1024, 256, 0, 0);

    float* xnext = (i == NLAYER - 1) ? (float*)d_out : x;
    ln_kernel<<<5440, 256, 0, stream>>>(x2, valb, ln2g + i * 256, ln2b + i * 256, xnext, nullptr);
  }
}
